// UpdateNodeInFrame_85744727097808
// MI455X (gfx1250) — matrix operand order repaired and checked by reading, not compiled, not run on hardware
//
#include <hip/hip_runtime.h>
#include <cstdint>
#include <cstddef>

typedef __bf16 bf16_t;
typedef bf16_t v16bf __attribute__((ext_vector_type(16)));
typedef float  v8f   __attribute__((ext_vector_type(8)));

#define NF   288          // 9*C
#define EPSL 1e-8f

// scales
#define S0       0.058925565098879f   // 1/sqrt(288)
#define S1       0.072168783648703f   // 1/sqrt(192)
#define S2       0.102062072615966f   // 1/sqrt(96)
#define SCP      0.176776695296637f   // 1/sqrt(32)
#define INVLAT   0.088388347648318f   // 1/sqrt(128)
#define C_OLD    0.894427190999916f   // 1/sqrt(1.25)
#define CMSG     0.141421356237310f   // 0.5/sqrt(1.25)/sqrt(10)

// packed bf16 weight offsets (elements) in workspace
#define OFF_B0   0        // [288][160]  (w0s|w01|w02)^T
#define OFF_B1P  46080    // [192][128]  (wr1_1|wi1_1|wr1_2|wi1_2)^T
#define OFF_B1M  70656    // [192][128]  (-wi1_1|wr1_1|-wi1_2|wr1_2)^T
#define OFF_B2P  95232    // [96][64]    (wr2_2|wi2_2)^T
#define OFF_B2M  101376   // [96][64]    (-wi2_2|wr2_2)^T
#define OFF_BENV 107520   // [128][96]   w_env^T
#define OFF_BP0  119808   // [32][32]
#define OFF_BP1  120832
#define OFF_BP2  121856
#define W_TOTAL  122880

__device__ __forceinline__ float wave_sum32(float x) {
#pragma unroll
  for (int m = 16; m >= 1; m >>= 1) x += __shfl_xor(x, m, 32);
  return x;
}
__device__ __forceinline__ float sigmoidf_(float x) { return 1.0f / (1.0f + expf(-x)); }

__device__ __forceinline__ void sh2f(float x, float y, float z, float* o) {
  const float SQ3 = 1.7320508075688772f;
  o[0] = SQ3 * z * x;
  o[1] = SQ3 * x * y;
  o[2] = 0.5f * (3.0f * y * y - 1.0f);
  o[3] = SQ3 * y * z;
  o[4] = 0.5f * SQ3 * (z * z - x * x);
}

// ---------------- WMMA fragment helpers (CDNA5 wave32 layouts) -------------
// A (16x32 bf16): lane<16 -> M=lane, halves {K+0..7, K+16..23}; lane>=16 -> M=lane-16, +8
__device__ __forceinline__ v16bf load_a_frag(const bf16_t* tile, int rowStride,
                                             int colBase, int lane) {
  const bf16_t* p = tile + (size_t)(lane & 15) * rowStride + colBase + ((lane & 16) ? 8 : 0);
  v16bf a;
  ((uint4*)&a)[0] = *(const uint4*)(p);
  ((uint4*)&a)[1] = *(const uint4*)(p + 16);
  return a;
}
// B (32x16 bf16): lane L -> N column nBase+(L&15), halves {K+0..7, K+16..23}; lane>=16 -> +8
__device__ __forceinline__ v16bf load_b_frag(const bf16_t* B, int ncols,
                                             int kBase, int nBase, int lane) {
  const bf16_t* p = B + (size_t)kBase * ncols + nBase + (lane & 15);
  const int kh = (lane & 16) ? 8 : 0;
  v16bf b;
#pragma unroll
  for (int s = 0; s < 16; ++s) b[s] = p[(size_t)(kh + (s < 8 ? s : s + 8)) * ncols];
  return b;
}
__device__ __forceinline__ v8f wmma_bf16(v16bf a, v16bf b, v8f c) {
  return __builtin_amdgcn_wmma_f32_16x16x32_bf16(false, a, false, b, (short)0, c,
                                                 false, false);
}
// D/C f32 16x16: lane<16 -> M=r, N=lane ; lane>=16 -> M=8+r, N=lane-16
__device__ __forceinline__ void store_d(float* tile, int ncols, int nBase, int lane,
                                        v8f acc, float scale) {
  int mb = (lane >> 4) * 8, nc = lane & 15;
#pragma unroll
  for (int r = 0; r < 8; ++r) tile[(size_t)(mb + r) * ncols + nBase + nc] = acc[r] * scale;
}

// ---------------- setup: quadrature constants (X, P2 = pinv(sh2(X))) -------
__global__ void prep_consts_kernel(float* __restrict__ P2 /*5x16*/,
                                   float* __restrict__ X /*16x3*/) {
  if (threadIdx.x != 0 || blockIdx.x != 0) return;
  float Xl[16][3], M[16][5];
  for (int k = 0; k < 16; ++k) {
    float zz = 1.0f - 2.0f * ((float)k + 0.5f) / 16.0f;
    float rr = sqrtf(fmaxf(0.0f, 1.0f - zz * zz));
    float ph = 2.3999632297286533f * (float)k;  // golden angle
    Xl[k][0] = rr * cosf(ph); Xl[k][1] = rr * sinf(ph); Xl[k][2] = zz;
    sh2f(Xl[k][0], Xl[k][1], Xl[k][2], M[k]);
  }
  // G = M^T M (5x5), invert via Gauss-Jordan, P2 = G^-1 M^T
  float A[5][10];
  for (int i = 0; i < 5; ++i)
    for (int j = 0; j < 10; ++j) {
      if (j < 5) {
        float s = 0.f;
        for (int k = 0; k < 16; ++k) s += M[k][i] * M[k][j];
        A[i][j] = s;
      } else A[i][j] = (j - 5 == i) ? 1.0f : 0.0f;
    }
  for (int col = 0; col < 5; ++col) {
    float ip = 1.0f / A[col][col];
    for (int j = 0; j < 10; ++j) A[col][j] *= ip;
    for (int r = 0; r < 5; ++r) if (r != col) {
      float f = A[r][col];
      for (int j = 0; j < 10; ++j) A[r][j] -= f * A[col][j];
    }
  }
  for (int m = 0; m < 5; ++m)
    for (int k = 0; k < 16; ++k) {
      float s = 0.f;
      for (int j = 0; j < 5; ++j) s += A[m][5 + j] * M[k][j];
      P2[m * 16 + k] = s;
    }
  for (int k = 0; k < 16; ++k)
    for (int i = 0; i < 3; ++i) X[k * 3 + i] = Xl[k][i];
}

// ---------------- pack all weights to bf16 K-major B matrices ---------------
__global__ void prep_weights_kernel(
    const float* __restrict__ w0s, const float* __restrict__ w01, const float* __restrict__ w02,
    const float* __restrict__ wr11, const float* __restrict__ wi11,
    const float* __restrict__ wr12, const float* __restrict__ wi12,
    const float* __restrict__ wr22, const float* __restrict__ wi22,
    const float* __restrict__ wp0, const float* __restrict__ wp1, const float* __restrict__ wp2,
    const float* __restrict__ w_env, bf16_t* __restrict__ Wb) {
  int i = blockIdx.x * blockDim.x + threadIdx.x;
  if (i >= W_TOTAL) return;
  float v = 0.f;
  if (i < OFF_B1P) {                       // B0 [288][160]
    int k = i / 160, n = i % 160;
    v = (n < 96) ? w0s[n * 288 + k] : (n < 128) ? w01[(n - 96) * 288 + k]
                                                : w02[(n - 128) * 288 + k];
  } else if (i < OFF_B1M) {                // B1p [192][128]
    int j = i - OFF_B1P; int k = j / 128, n = j % 128; int q = n >> 5, r = n & 31;
    v = (q == 0) ? wr11[r * 192 + k] : (q == 1) ? wi11[r * 192 + k]
        : (q == 2) ? wr12[r * 192 + k] : wi12[r * 192 + k];
  } else if (i < OFF_B2P) {                // B1m [192][128]
    int j = i - OFF_B1M; int k = j / 128, n = j % 128; int q = n >> 5, r = n & 31;
    v = (q == 0) ? -wi11[r * 192 + k] : (q == 1) ? wr11[r * 192 + k]
        : (q == 2) ? -wi12[r * 192 + k] : wr12[r * 192 + k];
  } else if (i < OFF_B2M) {                // B2p [96][64]
    int j = i - OFF_B2P; int k = j / 64, n = j % 64; int q = n >> 5, r = n & 31;
    v = q ? wi22[r * 96 + k] : wr22[r * 96 + k];
  } else if (i < OFF_BENV) {               // B2m [96][64]
    int j = i - OFF_B2M; int k = j / 64, n = j % 64; int q = n >> 5, r = n & 31;
    v = q ? wr22[r * 96 + k] : -wi22[r * 96 + k];
  } else if (i < OFF_BP0) {                // Benv [128][96]
    int j = i - OFF_BENV; int k = j / 96, n = j % 96;
    v = w_env[n * 128 + k];
  } else if (i < OFF_BP1) {
    int j = i - OFF_BP0; v = wp0[(j % 32) * 32 + (j / 32)];
  } else if (i < OFF_BP2) {
    int j = i - OFF_BP1; v = wp1[(j % 32) * 32 + (j / 32)];
  } else {
    int j = i - OFF_BP2; v = wp2[(j % 32) * 32 + (j / 32)];
  }
  Wb[i] = (bf16_t)v;
}

// ---------------- node normalization + output init --------------------------
__global__ void node_sln_kernel(const float* __restrict__ nf,
                                const float* __restrict__ g0, const float* __restrict__ b0,
                                const float* __restrict__ g1, const float* __restrict__ g2,
                                float* __restrict__ nfN, float* __restrict__ out, int N) {
  int n = blockIdx.x;
  if (n >= N) return;
  int c = threadIdx.x;  // 0..31
  const float* row = nf + (size_t)n * NF;
  float x0 = row[c];
  float x1[3], x2[5];
  float a1 = 0.f, a2 = 0.f;
#pragma unroll
  for (int m = 0; m < 3; ++m) { x1[m] = row[32 + 3 * c + m]; a1 += x1[m] * x1[m]; }
#pragma unroll
  for (int m = 0; m < 5; ++m) { x2[m] = row[128 + 5 * c + m]; a2 += x2[m] * x2[m]; }
  float sm = wave_sum32(x0);
  float sq = wave_sum32(x0 * x0);
  float s1s = wave_sum32(a1);
  float s2s = wave_sum32(a2);
  float mu = sm * (1.0f / 32.0f);
  float var = sq * (1.0f / 32.0f) - mu * mu;
  float inv0 = rsqrtf(var + EPSL);
  float sv = rsqrtf(0.5f * (s1s / 96.0f + s2s / 160.0f) + EPSL);
  float* orow = nfN + (size_t)n * NF;
  float* xrow = out + (size_t)n * NF;
  orow[c] = (x0 - mu) * inv0 * g0[c] + b0[c];
  xrow[c] = C_OLD * x0;
#pragma unroll
  for (int m = 0; m < 3; ++m) {
    orow[32 + 3 * c + m] = x1[m] * sv * g1[c];
    xrow[32 + 3 * c + m] = C_OLD * x1[m];
  }
#pragma unroll
  for (int m = 0; m < 5; ++m) {
    orow[128 + 5 * c + m] = x2[m] * sv * g2[c];
    xrow[128 + 5 * c + m] = C_OLD * x2[m];
  }
}

// ---------------- helper: rotate + scatter one 32-ch source row -------------
__device__ __forceinline__ void write_block(int e, int off, int c, float x0,
                                            const float* x1, const float* x2,
                                            const float* D1, const float* D2,
                                            bf16_t* in0, bf16_t* xp1, bf16_t* xm1,
                                            bf16_t* xp2, bf16_t* xm2) {
  float r1[3], r2[5];
#pragma unroll
  for (int m = 0; m < 3; ++m)
    r1[m] = D1[3 * m] * x1[0] + D1[3 * m + 1] * x1[1] + D1[3 * m + 2] * x1[2];
#pragma unroll
  for (int m = 0; m < 5; ++m) {
    float s = 0.f;
#pragma unroll
    for (int nn = 0; nn < 5; ++nn) s += D2[5 * m + nn] * x2[nn];
    r2[m] = s;
  }
  int col = off + c;
  in0[e * 288 + col]       = (bf16_t)x0;
  in0[e * 288 + 96 + col]  = (bf16_t)r1[1];
  in0[e * 288 + 192 + col] = (bf16_t)r2[2];
  xp1[e * 192 + col]       = (bf16_t)r1[2];
  xp1[e * 192 + 96 + col]  = (bf16_t)r2[3];
  xm1[e * 192 + col]       = (bf16_t)r1[0];
  xm1[e * 192 + 96 + col]  = (bf16_t)r2[1];
  xp2[e * 96 + col]        = (bf16_t)r2[4];
  xm2[e * 96 + col]        = (bf16_t)r2[0];
}

// ---------------- main edge kernel: 1 wave = 16-edge WMMA tile --------------
__global__ __launch_bounds__(32) void edge_kernel(
    const float* __restrict__ nfN, const float* __restrict__ edge_features,
    const float* __restrict__ edge_vector, const float* __restrict__ latents,
    const int* __restrict__ edge_index, int Eei,
    const int* __restrict__ active_edges, int EA,
    const float* __restrict__ g0e, const float* __restrict__ b0e,
    const float* __restrict__ g1e, const float* __restrict__ g2e,
    const bf16_t* __restrict__ Wb,
    const float* __restrict__ bp0, const float* __restrict__ b_env,
    const float* __restrict__ P2, const float* __restrict__ Xc,
    float* __restrict__ out) {
  const int lane = threadIdx.x;
  const int tile = blockIdx.x;

  __shared__ float sD1[16][9];
  __shared__ float sD2[16][25];
  __shared__ int sCtr[16], sNbr[16], sEdge[16], sAct[16];
  __shared__ union { bf16_t in0[16][288]; float y1[16][128]; } U1;
  __shared__ bf16_t sXp1[16][192], sXm1[16][192];
  __shared__ bf16_t sXp2[16][96], sXm2[16][96];
  __shared__ union {
    bf16_t lat[16][128];
    struct { bf16_t scal[16][32]; bf16_t g1[16][3][32]; } s;
  } U2;
  __shared__ bf16_t sG2[16][5][32];
  __shared__ float sY0[16][160];   // [y0(96)|y1_0(32)|y2_0(32)], pre-scaled
  __shared__ float sY2[16][64];    // [y2_p2|y2_m2]
  __shared__ float sEnv[16][96];

  // ---- Phase A: per-edge Wigner D1 (3x3) and D2 (5x5) ----
  if (lane < 16) {
    int eIdx = tile * 16 + lane;
    int act = (eIdx < EA) ? 1 : 0;
    int ae = act ? active_edges[eIdx] : 0;
    sAct[lane] = act;
    sEdge[lane] = ae;
    sCtr[lane] = edge_index[ae];
    sNbr[lane] = edge_index[Eei + ae];
    float vx = edge_vector[ae * 3 + 0];
    float vy = edge_vector[ae * 3 + 1];
    float vz = edge_vector[ae * 3 + 2];
    float nn = fmaxf(sqrtf(vx * vx + vy * vy + vz * vz), 1e-9f);
    float nx = vx / nn, ny = vy / nn, nz = vz / nn;
    float ax, ay, az;
    if (fabsf(nx) < 0.9f) { ax = 1.f; ay = 0.f; az = 0.f; }
    else { ax = 0.f; ay = 0.f; az = 1.f; }
    float dt = ax * nx + ay * ny + az * nz;
    float ux = ax - dt * nx, uy = ay - dt * ny, uz = az - dt * nz;
    float un = fmaxf(sqrtf(ux * ux + uy * uy + uz * uz), 1e-9f);
    ux /= un; uy /= un; uz /= un;
    float wx = uy * nz - uz * ny, wy = uz * nx - ux * nz, wz = ux * ny - uy * nx;
    float D1l[9] = {ux, uy, uz, nx, ny, nz, wx, wy, wz};
#pragma unroll
    for (int i = 0; i < 9; ++i) sD1[lane][i] = D1l[i];
    float D2l[25];
#pragma unroll
    for (int i = 0; i < 25; ++i) D2l[i] = 0.f;
    for (int k = 0; k < 16; ++k) {
      float X0 = Xc[3 * k], X1 = Xc[3 * k + 1], X2 = Xc[3 * k + 2];
      float px = D1l[0] * X0 + D1l[1] * X1 + D1l[2] * X2;
      float py = D1l[3] * X0 + D1l[4] * X1 + D1l[5] * X2;
      float pz = D1l[6] * X0 + D1l[7] * X1 + D1l[8] * X2;
      float s5[5];
      sh2f(px, py, pz, s5);
#pragma unroll
      for (int j = 0; j < 5; ++j) {
        float p = P2[j * 16 + k];
#pragma unroll
        for (int i = 0; i < 5; ++i) D2l[5 * i + j] += p * s5[i];
      }
    }
#pragma unroll
    for (int i = 0; i < 25; ++i) sD2[lane][i] = D2l[i];
  }
  __syncthreads();

  // ---- Phase B: build bf16 A tiles (gather/normalize/rotate), lane = channel
  const int c = lane;
  for (int e = 0; e < 16; ++e) {
    const int act = sAct[e];
    const int ae = sEdge[e];
    const float* D1 = sD1[e];
    const float* D2 = sD2[e];
    // center node (already normalized)
    {
      const float* row = nfN + (size_t)sCtr[e] * NF;
      float x1[3], x2[5];
      float x0 = row[c];
#pragma unroll
      for (int m = 0; m < 3; ++m) x1[m] = row[32 + 3 * c + m];
#pragma unroll
      for (int m = 0; m < 5; ++m) x2[m] = row[128 + 5 * c + m];
      write_block(e, 0, c, x0, x1, x2, D1, D2, &U1.in0[0][0], &sXp1[0][0],
                  &sXm1[0][0], &sXp2[0][0], &sXm2[0][0]);
    }
    // edge features: fused scalar-LN + tensor-norm
    {
      const float* row = edge_features + (size_t)ae * NF;
      float x0 = act ? row[c] : 0.f;
      float x1[3], x2[5];
      float a1 = 0.f, a2 = 0.f;
#pragma unroll
      for (int m = 0; m < 3; ++m) { x1[m] = act ? row[32 + 3 * c + m] : 0.f; a1 += x1[m] * x1[m]; }
#pragma unroll
      for (int m = 0; m < 5; ++m) { x2[m] = act ? row[128 + 5 * c + m] : 0.f; a2 += x2[m] * x2[m]; }
      float sm = wave_sum32(x0), sq = wave_sum32(x0 * x0);
      float s1s = wave_sum32(a1), s2s = wave_sum32(a2);
      float mu = sm * (1.0f / 32.0f);
      float var = sq * (1.0f / 32.0f) - mu * mu;
      float x0n = (x0 - mu) * rsqrtf(var + EPSL) * g0e[c] + b0e[c];
      float sv = rsqrtf(0.5f * (s1s / 96.0f + s2s / 160.0f) + EPSL);
      float x1n[3], x2n[5];
#pragma unroll
      for (int m = 0; m < 3; ++m) x1n[m] = x1[m] * sv * g1e[c];
#pragma unroll
      for (int m = 0; m < 5; ++m) x2n[m] = x2[m] * sv * g2e[c];
      write_block(e, 32, c, x0n, x1n, x2n, D1, D2, &U1.in0[0][0], &sXp1[0][0],
                  &sXm1[0][0], &sXp2[0][0], &sXm2[0][0]);
    }
    // neighbor node
    {
      const float* row = nfN + (size_t)sNbr[e] * NF;
      float x1[3], x2[5];
      float x0 = row[c];
#pragma unroll
      for (int m = 0; m < 3; ++m) x1[m] = row[32 + 3 * c + m];
#pragma unroll
      for (int m = 0; m < 5; ++m) x2[m] = row[128 + 5 * c + m];
      write_block(e, 64, c, x0, x1, x2, D1, D2, &U1.in0[0][0], &sXp1[0][0],
                  &sXm1[0][0], &sXp2[0][0], &sXm2[0][0]);
    }
    // latent row -> bf16
#pragma unroll
    for (int t = 0; t < 4; ++t) {
      int k = c + 32 * t;
      U2.lat[e][k] = (bf16_t)(act ? latents[(size_t)ae * 128 + k] : 0.f);
    }
  }
  __syncthreads();

  // ---- Phase C1: envelope GEMM (K=128 -> 96) ----
  for (int nt = 0; nt < 6; ++nt) {
    v8f acc = {0.f, 0.f, 0.f, 0.f, 0.f, 0.f, 0.f, 0.f};
    for (int kc = 0; kc < 4; ++kc)
      acc = wmma_bf16(load_a_frag(&U2.lat[0][0], 128, kc * 32, lane),
                      load_b_frag(Wb + OFF_BENV, 96, kc * 32, nt * 16, lane), acc);
    int mb = (lane >> 4) * 8, nc = lane & 15, col = nt * 16 + nc;
#pragma unroll
    for (int r = 0; r < 8; ++r) sEnv[mb + r][col] = acc[r] * INVLAT + b_env[col];
  }
  // ---- Phase C2: y0|y1_0|y2_0 GEMM (K=288 -> 160) ----
  for (int nt = 0; nt < 10; ++nt) {
    v8f acc = {0.f, 0.f, 0.f, 0.f, 0.f, 0.f, 0.f, 0.f};
    for (int kc = 0; kc < 9; ++kc)
      acc = wmma_bf16(load_a_frag(&U1.in0[0][0], 288, kc * 32, lane),
                      load_b_frag(Wb + OFF_B0, 160, kc * 32, nt * 16, lane), acc);
    store_d(&sY0[0][0], 160, nt * 16, lane, acc, S0);
  }
  __syncthreads();
  // ---- Phase C3: complex l=1 pair GEMMs (K=192 -> 128), reuses U1 as f32 y1
  for (int nt = 0; nt < 8; ++nt) {
    v8f acc = {0.f, 0.f, 0.f, 0.f, 0.f, 0.f, 0.f, 0.f};
    for (int kc = 0; kc < 6; ++kc)
      acc = wmma_bf16(load_a_frag(&sXp1[0][0], 192, kc * 32, lane),
                      load_b_frag(Wb + OFF_B1P, 128, kc * 32, nt * 16, lane), acc);
    for (int kc = 0; kc < 6; ++kc)
      acc = wmma_bf16(load_a_frag(&sXm1[0][0], 192, kc * 32, lane),
                      load_b_frag(Wb + OFF_B1M, 128, kc * 32, nt * 16, lane), acc);
    store_d(&U1.y1[0][0], 128, nt * 16, lane, acc, S1);
  }
  // ---- Phase C4: complex l=2,|m|=2 GEMMs (K=96 -> 64) ----
  for (int nt = 0; nt < 4; ++nt) {
    v8f acc = {0.f, 0.f, 0.f, 0.f, 0.f, 0.f, 0.f, 0.f};
    for (int kc = 0; kc < 3; ++kc)
      acc = wmma_bf16(load_a_frag(&sXp2[0][0], 96, kc * 32, lane),
                      load_b_frag(Wb + OFF_B2P, 64, kc * 32, nt * 16, lane), acc);
    for (int kc = 0; kc < 3; ++kc)
      acc = wmma_bf16(load_a_frag(&sXm2[0][0], 96, kc * 32, lane),
                      load_b_frag(Wb + OFF_B2M, 64, kc * 32, nt * 16, lane), acc);
    store_d(&sY2[0][0], 64, nt * 16, lane, acc, S2);
  }
  __syncthreads();

  // ---- Phase D: gating, back-rotation, rebuild bf16 A tiles ----
  for (int e = 0; e < 16; ++e) {
    float y0s = sY0[e][c];
    float g1v = sigmoidf_(sY0[e][32 + c]);
    float g2v = sigmoidf_(sY0[e][64 + c]);
    U2.s.scal[e][c] = (bf16_t)(y0s * sigmoidf_(y0s));  // silu
    float y1n[3] = {U1.y1[e][32 + c], sY0[e][96 + c], U1.y1[e][c]};
    float y2n[5] = {sY2[e][32 + c], U1.y1[e][96 + c], sY0[e][128 + c],
                    U1.y1[e][64 + c], sY2[e][c]};
#pragma unroll
    for (int m = 0; m < 3; ++m) {
      float s = 0.f;
#pragma unroll
      for (int nn = 0; nn < 3; ++nn) s += sD1[e][3 * nn + m] * y1n[nn];  // D1^T
      U2.s.g1[e][m][c] = (bf16_t)(s * g1v);
    }
#pragma unroll
    for (int m = 0; m < 5; ++m) {
      float s = 0.f;
#pragma unroll
      for (int nn = 0; nn < 5; ++nn) s += sD2[e][5 * nn + m] * y2n[nn];  // D2^T
      sG2[e][m][c] = (bf16_t)(s * g2v);
    }
  }
  __syncthreads();

  // ---- Phase E: 32x32 projections + envelope + atomic segment-sum ----
  const int mb = (lane >> 4) * 8, nc = lane & 15;
  // m0
  for (int nt = 0; nt < 2; ++nt) {
    v8f acc = {0.f, 0.f, 0.f, 0.f, 0.f, 0.f, 0.f, 0.f};
    acc = wmma_bf16(load_a_frag(&U2.s.scal[0][0], 32, 0, lane),
                    load_b_frag(Wb + OFF_BP0, 32, 0, nt * 16, lane), acc);
    int d = nt * 16 + nc;
#pragma unroll
    for (int r = 0; r < 8; ++r) {
      int e = mb + r;
      if (sAct[e]) {
        float val = (acc[r] * SCP + bp0[d]) * sEnv[e][d] * CMSG;
        atomicAdd(out + (size_t)sCtr[e] * NF + d, val);
      }
    }
  }
  // m1 (3 components)
  for (int m = 0; m < 3; ++m)
    for (int nt = 0; nt < 2; ++nt) {
      v8f acc = {0.f, 0.f, 0.f, 0.f, 0.f, 0.f, 0.f, 0.f};
      acc = wmma_bf16(load_a_frag(&U2.s.g1[0][0][0], 96, m * 32, lane),
                      load_b_frag(Wb + OFF_BP1, 32, 0, nt * 16, lane), acc);
      int d = nt * 16 + nc;
#pragma unroll
      for (int r = 0; r < 8; ++r) {
        int e = mb + r;
        if (sAct[e]) {
          float val = acc[r] * SCP * sEnv[e][32 + d] * CMSG;
          atomicAdd(out + (size_t)sCtr[e] * NF + 32 + 3 * d + m, val);
        }
      }
    }
  // m2 (5 components)
  for (int m = 0; m < 5; ++m)
    for (int nt = 0; nt < 2; ++nt) {
      v8f acc = {0.f, 0.f, 0.f, 0.f, 0.f, 0.f, 0.f, 0.f};
      acc = wmma_bf16(load_a_frag(&sG2[0][0][0], 160, m * 32, lane),
                      load_b_frag(Wb + OFF_BP2, 32, 0, nt * 16, lane), acc);
      int d = nt * 16 + nc;
#pragma unroll
      for (int r = 0; r < 8; ++r) {
        int e = mb + r;
        if (sAct[e]) {
          float val = acc[r] * SCP * sEnv[e][64 + d] * CMSG;
          atomicAdd(out + (size_t)sCtr[e] * NF + 128 + 5 * d + m, val);
        }
      }
    }
}

extern "C" void kernel_launch(void* const* d_in, const int* in_sizes, int n_in,
                              void* d_out, int out_size, void* d_ws, size_t ws_size,
                              hipStream_t stream) {
  const float* latents       = (const float*)d_in[0];
  const float* node_features = (const float*)d_in[1];
  const float* edge_features = (const float*)d_in[2];
  const float* edge_vector   = (const float*)d_in[3];
  const int*   edge_index    = (const int*)d_in[5];
  const int*   active_edges  = (const int*)d_in[6];
  const float* g0n = (const float*)d_in[7];
  const float* b0n = (const float*)d_in[8];
  const float* g1n = (const float*)d_in[9];
  const float* g2n = (const float*)d_in[10];
  const float* g0e = (const float*)d_in[11];
  const float* b0e = (const float*)d_in[12];
  const float* g1e = (const float*)d_in[13];
  const float* g2e = (const float*)d_in[14];
  const float* w0s  = (const float*)d_in[15];
  const float* w01  = (const float*)d_in[16];
  const float* w02  = (const float*)d_in[17];
  const float* wr11 = (const float*)d_in[18];
  const float* wi11 = (const float*)d_in[19];
  const float* wr12 = (const float*)d_in[20];
  const float* wi12 = (const float*)d_in[21];
  const float* wr22 = (const float*)d_in[22];
  const float* wi22 = (const float*)d_in[23];
  const float* wp0  = (const float*)d_in[24];
  const float* bp0  = (const float*)d_in[25];
  const float* wp1  = (const float*)d_in[26];
  const float* wp2  = (const float*)d_in[27];
  const float* w_env = (const float*)d_in[28];
  const float* b_env = (const float*)d_in[29];

  const int N   = in_sizes[1] / NF;
  const int EA  = in_sizes[6];       // active edge count
  const int Eei = in_sizes[5] / 2;   // edge_index row stride

  // workspace layout
  float*  P2  = (float*)d_ws;        // 80 floats
  float*  Xc  = P2 + 80;             // 48 floats
  float*  nfN = P2 + 128;            // N*288 floats
  bf16_t* Wb  = (bf16_t*)(nfN + (size_t)N * NF);  // W_TOTAL bf16

  prep_consts_kernel<<<1, 32, 0, stream>>>(P2, Xc);
  prep_weights_kernel<<<(W_TOTAL + 255) / 256, 256, 0, stream>>>(
      w0s, w01, w02, wr11, wi11, wr12, wi12, wr22, wi22, wp0, wp1, wp2, w_env, Wb);
  node_sln_kernel<<<N, 32, 0, stream>>>(node_features, g0n, b0n, g1n, g2n, nfN,
                                        (float*)d_out, N);
  int tiles = (EA + 15) / 16;
  edge_kernel<<<tiles, 32, 0, stream>>>(
      nfN, edge_features, edge_vector, latents, edge_index, Eei, active_edges, EA,
      g0e, b0e, g1e, g2e, Wb, bp0, b_env, P2, Xc, (float*)d_out);
  (void)n_in; (void)out_size; (void)ws_size;
}
